// AFNO2D_4131758539145
// MI455X (gfx1250) — matrix operand order repaired and checked by reading, not compiled, not run on hardware
//
#include <hip/hip_runtime.h>
#include <cstddef>

// ---------------------------------------------------------------------------
// AFNO2D for MI455X (gfx1250, wave32, WMMA).
// All transforms expressed as GEMMs:
//   fwd:  Xf = Fw(46x180) * x      (per b,h)        [f32 WMMA 16x16x4]
//         Y  = Fh(90x90 cplx) * Xf (per b,kcol)     [f32 WMMA 16x16x4]
//   MLP:  two 96x96 complex layers, bias+relu / bias+softshrink fused
//                                                   [f16 WMMA 16x16x32, f32 acc]
//   inv:  V  = Fh^-1 * Z,  y = G(180x46) * V + x    [f32 WMMA 16x16x4]
// Memory-bound (~1.6 GB traffic vs ~40 GFLOP), so fp32 matrix ops for the
// DFTs cost nothing; f16 matrix ops for the (well-scaled) MLP.
// ---------------------------------------------------------------------------

typedef __attribute__((ext_vector_type(2)))  float     v2f;
typedef __attribute__((ext_vector_type(8)))  float     v8f;
typedef __attribute__((ext_vector_type(16))) _Float16  v16h;

#define NB 8
#define BS 96
#define CC 768
#define HH 90
#define WW 180
#define FK 46      // kept rfft columns
#define NPOS 16560 // 4 * 46 * 90

__device__ __forceinline__ v8f wmma_f32(v2f a, v2f b, v8f c) {
  return __builtin_amdgcn_wmma_f32_16x16x4_f32(false, a, false, b, (short)0, c,
                                               false, false);
}
__device__ __forceinline__ v8f wmma_f16(v16h a, v16h b, v8f c) {
  return __builtin_amdgcn_wmma_f32_16x16x32_f16(false, a, false, b, (short)0, c,
                                                false, false);
}

// A-operand (16x32 f16): lane (l) holds row m=l&15, K = kbase..kbase+7 and
// kbase+16..kbase+23 where kbase = k0 + (l>>4)*8.  Source data is f32.
__device__ __forceinline__ v16h load_a16(const float* __restrict__ p, int kbase,
                                         float sgn) {
  v16h r;
#pragma unroll
  for (int j = 0; j < 8; ++j) {
    r[j]     = (_Float16)(sgn * p[kbase + j]);
    r[j + 8] = (_Float16)(sgn * p[kbase + 16 + j]);
  }
  return r;
}
// B-operand (32x16 f16): lane holds column n, K = kbase..kbase+7 and
// kbase+16..kbase+23, kbase = k0 + (l>>4)*8.
__device__ __forceinline__ v16h load_b16(const float* __restrict__ p, int ldb,
                                         int kbase, int col) {
  v16h r;
#pragma unroll
  for (int j = 0; j < 8; ++j) {
    r[j]     = (_Float16)p[(kbase + j) * ldb + col];
    r[j + 8] = (_Float16)p[(kbase + 16 + j) * ldb + col];
  }
  return r;
}

// ---------------------------------------------------------------------------
// DFT coefficient matrices (zero padded to WMMA tile multiples).
//   FwR/FwI : 48 x 180  (rows>=46 zero)          forward rfft over W, 1/sqrt(180)
//   FhR/FhI : 96 x 92   (rows>=90, cols>=90 zero) forward fft over H, 1/sqrt(90)
//   FiR/FiI : 96 x 92                              inverse fft over H
//   GR/GI   : 192 x 48  (rows>=180, cols>=46 zero) irfft over W with k>=1 doubling
// ---------------------------------------------------------------------------
__global__ void init_mats(float* FwR, float* FwI, float* FhR, float* FhI,
                          float* FiR, float* FiI, float* GR, float* GI) {
  const int t = blockIdx.x * blockDim.x + threadIdx.x;
  const int stride = gridDim.x * blockDim.x;
  const float PI2 = 6.28318530717958647692f;
  const float s180 = rsqrtf(180.f), s90 = rsqrtf(90.f);

  for (int i = t; i < 48 * 180; i += stride) {
    int k = i / 180, w = i % 180;
    float ang = PI2 * (float)((k * w) % 180) / 180.f;
    bool v = (k < FK);
    FwR[i] = v ? cosf(ang) * s180 : 0.f;
    FwI[i] = v ? -sinf(ang) * s180 : 0.f;
  }
  for (int i = t; i < 96 * 92; i += stride) {
    int hp = i / 92, h = i % 92;
    bool v = (hp < HH) && (h < HH);
    float ang = PI2 * (float)((hp * h) % 90) / 90.f;
    float c = v ? cosf(ang) * s90 : 0.f;
    float s = v ? sinf(ang) * s90 : 0.f;
    FhR[i] = c; FhI[i] = -s;   // e^{-i..}
    FiR[i] = c; FiI[i] = s;    // e^{+i..}
  }
  for (int i = t; i < 192 * 48; i += stride) {
    int w = i / 48, k = i % 48;
    bool v = (w < WW) && (k < FK);
    float alpha = (k == 0) ? 1.f : 2.f;
    float ang = PI2 * (float)((k * w) % 180) / 180.f;
    GR[i] = v ? alpha * cosf(ang) * s180 : 0.f;
    GI[i] = v ? -alpha * sinf(ang) * s180 : 0.f;
  }
}

// ---------------------------------------------------------------------------
// Stage 1: W-DFT forward.  Per bh: Xr/Xi[46x768] = Fw * x[180x768].
// grid (48, 3, 360), block 32 (one wave = one 16x16 tile, real+imag).
// ---------------------------------------------------------------------------
__global__ void k_wdft(const float* __restrict__ x, const float* __restrict__ FwR,
                       const float* __restrict__ FwI, float* __restrict__ Xr,
                       float* __restrict__ Xi) {
  const int l = threadIdx.x, lm = l & 15, hi = l >> 4;
  const int col = blockIdx.x * 16 + lm;
  const int mt = blockIdx.y;
  const int bh = blockIdx.z;
  const float* Bp = x + (size_t)bh * WW * CC;
  const int m = mt * 16 + lm;
  v8f accR = {}, accI = {};
  for (int k0 = 0; k0 < 180; k0 += 4) {
    int ka = k0 + hi * 2;
    v2f aR = *(const v2f*)(FwR + m * 180 + ka);
    v2f aI = *(const v2f*)(FwI + m * 180 + ka);
    v2f b;
    b.x = Bp[(size_t)ka * CC + col];
    b.y = Bp[(size_t)(ka + 1) * CC + col];
    accR = wmma_f32(aR, b, accR);
    accI = wmma_f32(aI, b, accI);
  }
#pragma unroll
  for (int v = 0; v < 8; ++v) {
    int mo = mt * 16 + hi * 8 + v;
    if (mo < FK) {
      size_t o = ((size_t)bh * FK + mo) * CC + col;
      Xr[o] = accR[v];
      Xi[o] = accI[v];
    }
  }
}

// ---------------------------------------------------------------------------
// Stage 2: complex H-DFT forward.  Per (b,k): Y[90x768] = Fh * X (complex).
// Output position-major: Yr[(b*46+k)*90 + h][768].  grid (48, 6, 184).
// ---------------------------------------------------------------------------
__global__ void k_hdft_fwd(const float* __restrict__ Xr, const float* __restrict__ Xi,
                           const float* __restrict__ AR, const float* __restrict__ AI,
                           float* __restrict__ Yr, float* __restrict__ Yi) {
  const int l = threadIdx.x, lm = l & 15, hi = l >> 4;
  const int col = blockIdx.x * 16 + lm;
  const int mt = blockIdx.y;
  const int bk = blockIdx.z;
  const int b = bk / FK, k = bk % FK;
  const int m = mt * 16 + lm;
  v8f accR = {}, accI = {};
  for (int k0 = 0; k0 < 92; k0 += 4) {
    int ka = k0 + hi * 2;
    v2f aR = *(const v2f*)(AR + m * 92 + ka);
    v2f aI = *(const v2f*)(AI + m * 92 + ka);
    v2f aIn; aIn.x = -aI.x; aIn.y = -aI.y;
    size_t r0 = ((size_t)(b * HH + ka) * FK + k) * CC + col;
    size_t r1 = ((size_t)(b * HH + ka + 1) * FK + k) * CC + col;
    v2f br, bi;
    br.x = Xr[r0]; br.y = Xr[r1];
    bi.x = Xi[r0]; bi.y = Xi[r1];
    accR = wmma_f32(aR, br, accR);
    accR = wmma_f32(aIn, bi, accR);
    accI = wmma_f32(aR, bi, accI);
    accI = wmma_f32(aI, br, accI);
  }
#pragma unroll
  for (int v = 0; v < 8; ++v) {
    int mo = mt * 16 + hi * 8 + v;
    if (mo < HH) {
      size_t o = ((size_t)bk * HH + mo) * CC + col;
      Yr[o] = accR[v];
      Yi[o] = accI[v];
    }
  }
}

// ---------------------------------------------------------------------------
// MLP layer 1: o1 = relu(W1a*xr - W1b*xi + b1[0],  W1a*xi + W1b*xr + b1[1])
// grid (6, 8, 1035): n-tile, channel-block, position-tile.  f16 WMMA, K=96.
// ---------------------------------------------------------------------------
__global__ void k_mlp1(const float* __restrict__ Ar, const float* __restrict__ Ai,
                       const float* __restrict__ w1, const float* __restrict__ b1,
                       float* __restrict__ Or, float* __restrict__ Oi) {
  const int l = threadIdx.x, lm = l & 15, hi = l >> 4;
  const int nt = blockIdx.x, nb = blockIdx.y, pt = blockIdx.z;
  const int col = nt * 16 + lm;
  const float* Wa = w1 + (size_t)nb * BS * BS;
  const float* Wb = w1 + (size_t)(NB + nb) * BS * BS;
  const int m = pt * 16 + lm;
  const float* ArP = Ar + (size_t)m * CC + nb * BS;
  const float* AiP = Ai + (size_t)m * CC + nb * BS;
  v8f accR = {}, accI = {};
#pragma unroll
  for (int s = 0; s < 3; ++s) {
    int ka = s * 32 + hi * 8;
    int kb = s * 32 + hi * 8;
    v16h ar  = load_a16(ArP, ka, 1.f);
    v16h ai  = load_a16(AiP, ka, 1.f);
    v16h ain = load_a16(AiP, ka, -1.f);
    v16h bwa = load_b16(Wa, BS, kb, col);
    v16h bwb = load_b16(Wb, BS, kb, col);
    accR = wmma_f16(ar, bwa, accR);
    accR = wmma_f16(ain, bwb, accR);
    accI = wmma_f16(ai, bwa, accI);
    accI = wmma_f16(ar, bwb, accI);
  }
  float biasR = b1[(size_t)nb * BS + col];
  float biasI = b1[(size_t)(NB + nb) * BS + col];
#pragma unroll
  for (int v = 0; v < 8; ++v) {
    int mo = pt * 16 + hi * 8 + v;
    size_t o = (size_t)mo * CC + nb * BS + col;
    float r  = accR[v] + biasR;
    float im = accI[v] + biasI;
    Or[o] = r > 0.f ? r : 0.f;
    Oi[o] = im > 0.f ? im : 0.f;
  }
}

// ---------------------------------------------------------------------------
// MLP layer 2: o2 = softshrink(W2a*o1r - W2b*o1i + b2[0], ...)
// ---------------------------------------------------------------------------
__global__ void k_mlp2(const float* __restrict__ Ar, const float* __restrict__ Ai,
                       const float* __restrict__ w2, const float* __restrict__ b2,
                       float* __restrict__ Or, float* __restrict__ Oi) {
  const int l = threadIdx.x, lm = l & 15, hi = l >> 4;
  const int nt = blockIdx.x, nb = blockIdx.y, pt = blockIdx.z;
  const int col = nt * 16 + lm;
  const float* Wa = w2 + (size_t)nb * BS * BS;
  const float* Wb = w2 + (size_t)(NB + nb) * BS * BS;
  const int m = pt * 16 + lm;
  const float* ArP = Ar + (size_t)m * CC + nb * BS;
  const float* AiP = Ai + (size_t)m * CC + nb * BS;
  v8f accR = {}, accI = {};
#pragma unroll
  for (int s = 0; s < 3; ++s) {
    int ka = s * 32 + hi * 8;
    int kb = s * 32 + hi * 8;
    v16h ar  = load_a16(ArP, ka, 1.f);
    v16h ai  = load_a16(AiP, ka, 1.f);
    v16h ain = load_a16(AiP, ka, -1.f);
    v16h bwa = load_b16(Wa, BS, kb, col);
    v16h bwb = load_b16(Wb, BS, kb, col);
    accR = wmma_f16(ar, bwa, accR);
    accR = wmma_f16(ain, bwb, accR);
    accI = wmma_f16(ai, bwa, accI);
    accI = wmma_f16(ar, bwb, accI);
  }
  float biasR = b2[(size_t)nb * BS + col];
  float biasI = b2[(size_t)(NB + nb) * BS + col];
  const float lam = 0.01f;
#pragma unroll
  for (int v = 0; v < 8; ++v) {
    int mo = pt * 16 + hi * 8 + v;
    size_t o = (size_t)mo * CC + nb * BS + col;
    float tr = accR[v] + biasR;
    float ti = accI[v] + biasI;
    Or[o] = (tr > lam) ? tr - lam : ((tr < -lam) ? tr + lam : 0.f);
    Oi[o] = (ti > lam) ? ti - lam : ((ti < -lam) ? ti + lam : 0.f);
  }
}

// ---------------------------------------------------------------------------
// Stage 4: complex H-DFT inverse.  Per (b,k): V = Fh^-1 * Z.
// Z rows position-major [bk*90+h]; output Vr[(b*90+h)*46 + k][768].
// ---------------------------------------------------------------------------
__global__ void k_hdft_inv(const float* __restrict__ Zr, const float* __restrict__ Zi,
                           const float* __restrict__ AR, const float* __restrict__ AI,
                           float* __restrict__ Vr, float* __restrict__ Vi) {
  const int l = threadIdx.x, lm = l & 15, hi = l >> 4;
  const int col = blockIdx.x * 16 + lm;
  const int mt = blockIdx.y;
  const int bk = blockIdx.z;
  const int b = bk / FK, k = bk % FK;
  const int m = mt * 16 + lm;
  v8f accR = {}, accI = {};
  for (int k0 = 0; k0 < 92; k0 += 4) {
    int ka = k0 + hi * 2;
    v2f aR = *(const v2f*)(AR + m * 92 + ka);
    v2f aI = *(const v2f*)(AI + m * 92 + ka);
    v2f aIn; aIn.x = -aI.x; aIn.y = -aI.y;
    size_t r0 = ((size_t)bk * HH + ka) * CC + col;
    size_t r1 = r0 + CC;
    v2f br, bi;
    br.x = Zr[r0]; br.y = Zr[r1];
    bi.x = Zi[r0]; bi.y = Zi[r1];
    accR = wmma_f32(aR, br, accR);
    accR = wmma_f32(aIn, bi, accR);
    accI = wmma_f32(aR, bi, accI);
    accI = wmma_f32(aI, br, accI);
  }
#pragma unroll
  for (int v = 0; v < 8; ++v) {
    int mo = mt * 16 + hi * 8 + v;
    if (mo < HH) {
      size_t o = ((size_t)(b * HH + mo) * FK + k) * CC + col;
      Vr[o] = accR[v];
      Vi[o] = accI[v];
    }
  }
}

// ---------------------------------------------------------------------------
// Stage 5: inverse W-DFT + residual.  Per bh: y = G_R*Vr + G_I*Vi + x.
// grid (48, 12, 360).
// ---------------------------------------------------------------------------
__global__ void k_iwdft(const float* __restrict__ Vr, const float* __restrict__ Vi,
                        const float* __restrict__ GR, const float* __restrict__ GI,
                        const float* __restrict__ x, float* __restrict__ y) {
  const int l = threadIdx.x, lm = l & 15, hi = l >> 4;
  const int col = blockIdx.x * 16 + lm;
  const int mt = blockIdx.y;
  const int bh = blockIdx.z;
  const int m = mt * 16 + lm;
  v8f acc = {};
  for (int k0 = 0; k0 < 48; k0 += 4) {
    int ka = k0 + hi * 2;
    v2f aR = *(const v2f*)(GR + m * 48 + ka);
    v2f aI = *(const v2f*)(GI + m * 48 + ka);
    size_t r0 = ((size_t)bh * FK + ka) * CC + col;
    size_t r1 = r0 + CC;
    v2f br, bi;
    br.x = Vr[r0]; br.y = Vr[r1];
    bi.x = Vi[r0]; bi.y = Vi[r1];
    acc = wmma_f32(aR, br, acc);
    acc = wmma_f32(aI, bi, acc);
  }
#pragma unroll
  for (int v = 0; v < 8; ++v) {
    int w = mt * 16 + hi * 8 + v;
    if (w < WW) {
      size_t o = ((size_t)bh * WW + w) * CC + col;
      y[o] = acc[v] + x[o];
    }
  }
}

// ---------------------------------------------------------------------------
extern "C" void kernel_launch(void* const* d_in, const int* in_sizes, int n_in,
                              void* d_out, int out_size, void* d_ws, size_t ws_size,
                              hipStream_t stream) {
  (void)in_sizes; (void)n_in; (void)out_size; (void)ws_size;
  const float* x  = (const float*)d_in[0];
  const float* w1 = (const float*)d_in[1];
  const float* b1 = (const float*)d_in[2];
  const float* w2 = (const float*)d_in[3];
  const float* b2 = (const float*)d_in[4];
  float* out = (float*)d_out;

  // Workspace carve-up (floats).  Big buffers are ping-ponged:
  //  BufA: stage1 out (Xr/Xi) -> stage2 in; mlp1 out -> mlp2 in; stage4 out -> stage5 in
  //  BufB: stage2 out (pos-major) -> mlp1 in; mlp2 out (Z) -> stage4 in
  float* p = (float*)d_ws;
  float* FwR = p; p += 48 * 180;
  float* FwI = p; p += 48 * 180;
  float* FhR = p; p += 96 * 92;
  float* FhI = p; p += 96 * 92;
  float* FiR = p; p += 96 * 92;
  float* FiI = p; p += 96 * 92;
  float* GR  = p; p += 192 * 48;
  float* GI  = p; p += 192 * 48;
  const size_t SA = (size_t)16652 * CC;  // 362*46 rows (+pad for K over-read)
  const size_t SB = (size_t)16562 * CC;  // 16560 positions + 2 pad rows
  float* BufAr = p; p += SA;
  float* BufAi = p; p += SA;
  float* BufBr = p; p += SB;
  float* BufBi = p; /* p += SB; */

  init_mats<<<64, 256, 0, stream>>>(FwR, FwI, FhR, FhI, FiR, FiI, GR, GI);

  k_wdft    <<<dim3(48,  3, 360), 32, 0, stream>>>(x, FwR, FwI, BufAr, BufAi);
  k_hdft_fwd<<<dim3(48,  6, 184), 32, 0, stream>>>(BufAr, BufAi, FhR, FhI, BufBr, BufBi);
  k_mlp1    <<<dim3(6,   8, 1035), 32, 0, stream>>>(BufBr, BufBi, w1, b1, BufAr, BufAi);
  k_mlp2    <<<dim3(6,   8, 1035), 32, 0, stream>>>(BufAr, BufAi, w2, b2, BufBr, BufBi);
  k_hdft_inv<<<dim3(48,  6, 184), 32, 0, stream>>>(BufBr, BufBi, FiR, FiI, BufAr, BufAi);
  k_iwdft   <<<dim3(48, 12, 360), 32, 0, stream>>>(BufAr, BufAi, GR, GI, x, out);
}
